// ViewControlInjector_21526376087941
// MI455X (gfx1250) — matrix operand order repaired and checked by reading, not compiled, not run on hardware
//
#include <hip/hip_runtime.h>
#include <hip/hip_bf16.h>

typedef __attribute__((ext_vector_type(16))) __bf16 v16bf;
typedef __attribute__((ext_vector_type(8)))  float  v8f;
typedef __attribute__((ext_vector_type(4)))  int    v4i;

#define NTOK 2640
#define SPAT 880
#define DHID 1024
#define CIMG 3072
#define NHEAD 16
#define DHEAD 64

#if defined(__has_builtin)
#if __has_builtin(__builtin_amdgcn_global_load_async_to_lds_b128)
#define HAS_ASYNC_LDS 1
#endif
#endif
#ifndef HAS_ASYNC_LDS
#define HAS_ASYNC_LDS 0
#endif

__device__ inline unsigned short f32_to_bf16(float f) {
    unsigned int u = __builtin_bit_cast(unsigned int, f);
    unsigned int r = (u + 0x7FFFu + ((u >> 16) & 1u)) >> 16;
    return (unsigned short)r;
}
__device__ inline __bf16 us_to_bf(unsigned short h) {
    return __builtin_bit_cast(__bf16, h);
}
__device__ inline float gelu_tanh(float x) {
    float x3 = x * x * x;
    float u = 0.7978845608028654f * (x + 0.044715f * x3);
    return 0.5f * x * (1.0f + tanhf(u));
}

// 16B global -> LDS copy: async (ASYNCcnt-tracked, VGPR-bypassing).
__device__ inline void cp_b128(const void* g, void* l) {
#if HAS_ASYNC_LDS
    __builtin_amdgcn_global_load_async_to_lds_b128(
        (__attribute__((address_space(1))) v4i*)g,
        (__attribute__((address_space(3))) v4i*)l, 0, 0);
#else
    *(uint4*)l = *(const uint4*)g;
#endif
}
__device__ inline void cp_wait() {
#if HAS_ASYNC_LDS
#if __has_builtin(__builtin_amdgcn_s_wait_asynccnt)
    __builtin_amdgcn_s_wait_asynccnt(0);
#else
    asm volatile("s_wait_asynccnt 0" ::: "memory");
#endif
#endif
}

// ---------------------------------------------------------------------------
// elementwise f32 -> bf16 conversion, 4 elems/thread, packed stores
// ---------------------------------------------------------------------------
__global__ __launch_bounds__(256) void f2bf_kernel(const float* __restrict__ in,
                                                   unsigned short* __restrict__ out,
                                                   int n) {
    int i = (blockIdx.x * 256 + threadIdx.x) * 4;
    if (i < n) {
        float4 v = *(const float4*)(in + i);
        uint2 p;
        p.x = (unsigned int)f32_to_bf16(v.x) | ((unsigned int)f32_to_bf16(v.y) << 16);
        p.y = (unsigned int)f32_to_bf16(v.z) | ((unsigned int)f32_to_bf16(v.w) << 16);
        *(uint2*)(out + i) = p;
    }
}

// ---------------------------------------------------------------------------
// bias3[t][d] = b1[d] + sum_j gm[t][j] * w1[3072+j][d]
// ---------------------------------------------------------------------------
__global__ __launch_bounds__(256) void bias_prep_kernel(const float* __restrict__ cond,
                                                        const float* __restrict__ w1,
                                                        const float* __restrict__ b1,
                                                        float* __restrict__ bias3) {
    int t = blockIdx.x; // 0..2
    float gm[24];
    for (int p = 0; p < 12; ++p)
        for (int c = 0; c < 2; ++c) {
            int src = t * 4 + p - 8;
            gm[p * 2 + c] = (src >= 0) ? cond[src * 2 + c] : 0.0f;
        }
    for (int i = 0; i < 4; ++i) {
        int d = threadIdx.x + i * 256;
        float acc = b1[d];
        for (int j = 0; j < 24; ++j)
            acc += gm[j] * w1[(size_t)(CIMG + j) * DHID + d];
        bias3[t * DHID + d] = acc;
    }
}

// ---------------------------------------------------------------------------
// bf16 WMMA GEMM: C(M,N) = A(M,K) @ W(K,N)
// 256 threads (8 waves), 128x128 C tile, K-step 64, double-buffered LDS with
// async global->LDS tile fills. Per K-chunk all fragments (4 A + 2 B) are
// loaded before the 8-WMMA burst so LDS latency overlaps matrix math.
// EPI 0: += bias3[row%3][col], GELU, store bf16
// EPI 1: += bias[col], store f32
// EPI 2: += bias[col] + resid[row*N+col], store f32 (ROWMAP remaps A rows)
// ---------------------------------------------------------------------------
template <int EPI, int ROWMAP>
__global__ __launch_bounds__(256) void gemm_bf16_kernel(
    const unsigned short* __restrict__ A, const unsigned short* __restrict__ W,
    const float* __restrict__ bias, const float* __restrict__ extra,
    void* __restrict__ outp, int M, int N, int K) {
    __shared__ unsigned short As[2][128 * 64];
    __shared__ unsigned short Ws[2][64 * 128];

    const int tid = threadIdx.x;
    const int wave = tid >> 5, lane = tid & 31;
    const int lane16 = lane & 15, hi = lane >> 4;
    const int wr = wave >> 2, wc = wave & 3;
    const int nb = blockIdx.x * 128;
    const int mb = blockIdx.y * 128;

    v8f acc[4][2];
    for (int m = 0; m < 4; ++m)
        for (int n = 0; n < 2; ++n)
            for (int r = 0; r < 8; ++r) acc[m][n][r] = 0.0f;

    auto issue = [&](int buf, int kb) {
        // A tile: 128 rows x 64 k  = 1024 x 16B chunks (4 per thread)
        for (int i = 0; i < 4; ++i) {
            int idx = tid * 4 + i;
            int row = idx >> 3, c4 = idx & 7;
            int gr = mb + row;
            if (gr >= M) gr = M - 1; // clamped rows are never stored
            int ar = gr;
            if (ROWMAP) { int s = gr % SPAT, t = gr / SPAT; ar = s * 3 + t; }
            cp_b128(A + (size_t)ar * K + kb + c4 * 8, &As[buf][row * 64 + c4 * 8]);
        }
        // W tile: 64 rows x 128 cols = 1024 x 16B chunks (4 per thread)
        for (int i = 0; i < 4; ++i) {
            int idx = tid * 4 + i;
            int row = idx >> 4, c4 = idx & 15;
            cp_b128(W + (size_t)(kb + row) * N + nb + c4 * 8, &Ws[buf][row * 128 + c4 * 8]);
        }
    };

    issue(0, 0);
    int buf = 0;
    for (int kb = 0; kb < K; kb += 64) {
        cp_wait();
        __syncthreads();
        if (kb + 64 < K) issue(buf ^ 1, kb + 64);
        for (int c = 0; c < 2; ++c) {
            // hoist ALL fragment loads ahead of the WMMA burst
            v16bf af[4], bfr[2];
            for (int m = 0; m < 4; ++m)
                for (int e = 0; e < 16; ++e) {
                    int k = c * 32 + e + (hi ? 16 : 0);
                    af[m][e] = us_to_bf(As[buf][(wr * 64 + m * 16 + lane16) * 64 + k]);
                }
            for (int n = 0; n < 2; ++n)
                for (int e = 0; e < 16; ++e) {
                    int k = c * 32 + e + (hi ? 16 : 0);
                    bfr[n][e] = us_to_bf(Ws[buf][k * 128 + wc * 32 + n * 16 + lane16]);
                }
            for (int n = 0; n < 2; ++n)
                for (int m = 0; m < 4; ++m)
                    acc[m][n] = __builtin_amdgcn_wmma_f32_16x16x32_bf16(
                        false, af[m], false, bfr[n], (short)0, acc[m][n], false, false);
        }
        buf ^= 1;
    }

    for (int m = 0; m < 4; ++m)
        for (int n = 0; n < 2; ++n) {
            int col = nb + wc * 32 + n * 16 + lane16;
            for (int r = 0; r < 8; ++r) {
                int row = mb + wr * 64 + m * 16 + r + (hi ? 8 : 0);
                if (row >= M) continue;
                float v = acc[m][n][r];
                if (EPI == 0) {
                    int t = row % 3;
                    v += extra[t * DHID + col];
                    v = gelu_tanh(v);
                    ((unsigned short*)outp)[(size_t)row * N + col] = f32_to_bf16(v);
                } else if (EPI == 1) {
                    v += bias[col];
                    ((float*)outp)[(size_t)row * N + col] = v;
                } else {
                    v += bias[col];
                    v += extra[(size_t)row * N + col];
                    ((float*)outp)[(size_t)row * N + col] = v;
                }
            }
        }
}

// ---------------------------------------------------------------------------
// row-wise LayerNorm (1024 wide), f32 in -> bf16 out
// ---------------------------------------------------------------------------
__global__ __launch_bounds__(256) void ln_kernel(const float* __restrict__ h2,
                                                 const float* __restrict__ g,
                                                 const float* __restrict__ b,
                                                 unsigned short* __restrict__ out) {
    int row = blockIdx.x;
    const float* x = h2 + (size_t)row * DHID;
    float v4[4], s = 0.0f, s2 = 0.0f;
    for (int i = 0; i < 4; ++i) {
        float v = x[threadIdx.x + i * 256];
        v4[i] = v; s += v; s2 += v * v;
    }
    for (int m = 1; m < 32; m <<= 1) {
        s += __shfl_xor(s, m, 32);
        s2 += __shfl_xor(s2, m, 32);
    }
    __shared__ float ws1[8], ws2[8];
    if ((threadIdx.x & 31) == 0) { ws1[threadIdx.x >> 5] = s; ws2[threadIdx.x >> 5] = s2; }
    __syncthreads();
    float ts = 0.0f, ts2 = 0.0f;
    for (int i = 0; i < 8; ++i) { ts += ws1[i]; ts2 += ws2[i]; }
    float mean = ts * (1.0f / DHID);
    float var = ts2 * (1.0f / DHID) - mean * mean;
    float inv = rsqrtf(var + 1e-5f);
    for (int i = 0; i < 4; ++i) {
        int d = threadIdx.x + i * 256;
        float y = (v4[i] - mean) * inv * g[d] + b[d];
        out[(size_t)row * DHID + d] = f32_to_bf16(y);
    }
}

// ---------------------------------------------------------------------------
// QKV post: split qkv (token,[3][16][64]), RMS-norm q/k over 64, write
// head-major bf16 q/k/v: [head][token][64]
// ---------------------------------------------------------------------------
__global__ __launch_bounds__(64) void qkv_post_kernel(const float* __restrict__ qkvf,
                                                      const float* __restrict__ qn,
                                                      const float* __restrict__ kn,
                                                      unsigned short* __restrict__ qb,
                                                      unsigned short* __restrict__ kb,
                                                      unsigned short* __restrict__ vb) {
    int head = blockIdx.x, n = blockIdx.y;
    int d = threadIdx.x;
    const float* base = qkvf + (size_t)n * (3 * DHID);
    float q = base[head * DHEAD + d];
    float k = base[DHID + head * DHEAD + d];
    float v = base[2 * DHID + head * DHEAD + d];
    float qs = q * q, ks = k * k;
    for (int m = 1; m < 32; m <<= 1) {
        qs += __shfl_xor(qs, m, 32);
        ks += __shfl_xor(ks, m, 32);
    }
    __shared__ float sq[2], sk[2];
    if ((threadIdx.x & 31) == 0) { sq[threadIdx.x >> 5] = qs; sk[threadIdx.x >> 5] = ks; }
    __syncthreads();
    float qi = rsqrtf((sq[0] + sq[1]) * (1.0f / DHEAD) + 1e-6f);
    float ki = rsqrtf((sk[0] + sk[1]) * (1.0f / DHEAD) + 1e-6f);
    size_t o = (size_t)head * NTOK * DHEAD + (size_t)n * DHEAD + d;
    qb[o] = f32_to_bf16(q * qi * qn[d]);
    kb[o] = f32_to_bf16(k * ki * kn[d]);
    vb[o] = f32_to_bf16(v);
}

// ---------------------------------------------------------------------------
// flash attention: grid (qtiles=42, heads=16), block 128 (4 waves).
// Each wave: 16 query rows; streams K/V in 64-key tiles through LDS (async
// fills); QK^T and PV via v_wmma_f32_16x16x32_bf16; online softmax per row.
// All fragment loads for a K-chunk are hoisted ahead of the WMMA burst.
// ---------------------------------------------------------------------------
__global__ __launch_bounds__(128) void flash_attn_kernel(const unsigned short* __restrict__ qb,
                                                         const unsigned short* __restrict__ kbuf,
                                                         const unsigned short* __restrict__ vbuf,
                                                         unsigned short* __restrict__ ob) {
    __shared__ unsigned short Qs[64 * 64];
    __shared__ unsigned short Ks[64 * 64];
    __shared__ unsigned short Vs[64 * 64];
    __shared__ unsigned short Ps[4][16 * 64];

    const int head = blockIdx.y;
    const int q0 = blockIdx.x * 64;
    const int tid = threadIdx.x;
    const int wave = tid >> 5, lane = tid & 31;
    const int lane16 = lane & 15, hi = lane >> 4;

    const unsigned short* Qh = qb + (size_t)head * NTOK * DHEAD;
    const unsigned short* Kh = kbuf + (size_t)head * NTOK * DHEAD;
    const unsigned short* Vh = vbuf + (size_t)head * NTOK * DHEAD;

    // Q tile 64x64: 512 x 16B chunks (4 per thread), OOB rows clamped
    for (int i = 0; i < 4; ++i) {
        int idx = tid * 4 + i;
        int row = idx >> 3, c4 = idx & 7;
        int g = q0 + row;
        if (g >= NTOK) g = NTOK - 1;
        cp_b128(Qh + (size_t)g * DHEAD + c4 * 8, &Qs[row * 64 + c4 * 8]);
    }
    cp_wait();
    __syncthreads();

    v16bf qf[2];
    for (int c = 0; c < 2; ++c)
        for (int e = 0; e < 16; ++e) {
            int k = c * 32 + e + (hi ? 16 : 0);
            qf[c][e] = us_to_bf(Qs[(wave * 16 + lane16) * 64 + k]);
        }

    v8f O[4];
    float mrow[8], lrow[8];
    for (int g = 0; g < 4; ++g)
        for (int r = 0; r < 8; ++r) O[g][r] = 0.0f;
    for (int r = 0; r < 8; ++r) { mrow[r] = -1e30f; lrow[r] = 0.0f; }
    const float scale = 0.125f; // 1/sqrt(64)

    for (int kb0 = 0; kb0 < NTOK; kb0 += 64) {
        __syncthreads();
        for (int i = 0; i < 4; ++i) {
            int idx = tid * 4 + i;
            int row = idx >> 3, c4 = idx & 7;
            int g = kb0 + row;
            if (g >= NTOK) g = NTOK - 1; // clamped keys masked below
            cp_b128(Kh + (size_t)g * DHEAD + c4 * 8, &Ks[row * 64 + c4 * 8]);
            cp_b128(Vh + (size_t)g * DHEAD + c4 * 8, &Vs[row * 64 + c4 * 8]);
        }
        cp_wait();
        __syncthreads();

        // S = Q @ K^T (16 queries x 64 keys per wave); per chunk load all 4
        // K-fragments, then 4 WMMAs
        v8f S[4];
        for (int j = 0; j < 4; ++j)
            for (int r = 0; r < 8; ++r) S[j][r] = 0.0f;
        for (int c = 0; c < 2; ++c) {
            v16bf bfr[4];
            for (int j = 0; j < 4; ++j)
                for (int e = 0; e < 16; ++e) {
                    int kd = c * 32 + e + (hi ? 16 : 0);
                    bfr[j][e] = us_to_bf(Ks[(j * 16 + lane16) * 64 + kd]);
                }
            for (int j = 0; j < 4; ++j)
                S[j] = __builtin_amdgcn_wmma_f32_16x16x32_bf16(
                    false, qf[c], false, bfr[j], (short)0, S[j], false, false);
        }
        // scale + OOB key mask
        for (int j = 0; j < 4; ++j) {
            bool oob = (kb0 + j * 16 + lane16) >= NTOK;
            for (int r = 0; r < 8; ++r) S[j][r] = oob ? -1e30f : S[j][r] * scale;
        }
        // online softmax per row
        float alpha[8];
        for (int r = 0; r < 8; ++r) {
            float mx = fmaxf(fmaxf(S[0][r], S[1][r]), fmaxf(S[2][r], S[3][r]));
            for (int m = 1; m < 16; m <<= 1) mx = fmaxf(mx, __shfl_xor(mx, m, 16));
            float mnew = fmaxf(mrow[r], mx);
            alpha[r] = __expf(mrow[r] - mnew);
            mrow[r] = mnew;
            float rs = 0.0f;
            for (int j = 0; j < 4; ++j) {
                float p = __expf(S[j][r] - mnew);
                S[j][r] = p;
                rs += p;
            }
            for (int m = 1; m < 16; m <<= 1) rs += __shfl_xor(rs, m, 16);
            lrow[r] = lrow[r] * alpha[r] + rs;
        }
        for (int g = 0; g < 4; ++g)
            for (int r = 0; r < 8; ++r) O[g][r] *= alpha[r];
        // P: C-layout -> A-layout via wave-private LDS
        for (int j = 0; j < 4; ++j)
            for (int r = 0; r < 8; ++r) {
                int row = r + (hi ? 8 : 0);
                Ps[wave][row * 64 + j * 16 + lane16] = f32_to_bf16(S[j][r]);
            }
        // O += P @ V; per chunk load P-fragment + all 4 V-fragments, then 4 WMMAs
        for (int c = 0; c < 2; ++c) {
            v16bf pf;
            for (int e = 0; e < 16; ++e) {
                int k = c * 32 + e + (hi ? 16 : 0);
                pf[e] = us_to_bf(Ps[wave][lane16 * 64 + k]);
            }
            v16bf vfr[4];
            for (int g = 0; g < 4; ++g)
                for (int e = 0; e < 16; ++e) {
                    int kk = c * 32 + e + (hi ? 16 : 0);
                    vfr[g][e] = us_to_bf(Vs[kk * 64 + g * 16 + lane16]);
                }
            for (int g = 0; g < 4; ++g)
                O[g] = __builtin_amdgcn_wmma_f32_16x16x32_bf16(
                    false, pf, false, vfr[g], (short)0, O[g], false, false);
        }
    }

    for (int g = 0; g < 4; ++g)
        for (int r = 0; r < 8; ++r) {
            int row = q0 + wave * 16 + r + (hi ? 8 : 0);
            if (row < NTOK) {
                float val = O[g][r] / lrow[r];
                ob[(size_t)row * DHID + head * DHEAD + g * 16 + lane16] = f32_to_bf16(val);
            }
        }
}

// ---------------------------------------------------------------------------
extern "C" void kernel_launch(void* const* d_in, const int* in_sizes, int n_in,
                              void* d_out, int out_size, void* d_ws, size_t ws_size,
                              hipStream_t stream) {
    const float* x      = (const float*)d_in[0];
    const float* cond   = (const float*)d_in[1];
    const float* w1     = (const float*)d_in[2];
    const float* b1     = (const float*)d_in[3];
    const float* w2     = (const float*)d_in[4];
    const float* b2     = (const float*)d_in[5];
    const float* ln_g   = (const float*)d_in[6];
    const float* ln_b   = (const float*)d_in[7];
    const float* wqkv   = (const float*)d_in[8];
    const float* bqkv   = (const float*)d_in[9];
    const float* qn_w   = (const float*)d_in[10];
    const float* kn_w   = (const float*)d_in[11];
    const float* wproj  = (const float*)d_in[12];
    const float* bproj  = (const float*)d_in[13];
    float* out = (float*)d_out;

    char* ws = (char*)d_ws;
    size_t off = 0;
    auto alloc = [&](size_t bytes) { size_t o = off; off += (bytes + 255) & ~(size_t)255; return o; };

    unsigned short* xb     = (unsigned short*)(ws + alloc((size_t)NTOK * CIMG * 2));
    unsigned short* w1b    = (unsigned short*)(ws + alloc((size_t)CIMG * DHID * 2));
    unsigned short* w2b    = (unsigned short*)(ws + alloc((size_t)DHID * DHID * 2));
    unsigned short* wqkvb  = (unsigned short*)(ws + alloc((size_t)DHID * 3 * DHID * 2));
    unsigned short* wprojb = (unsigned short*)(ws + alloc((size_t)DHID * CIMG * 2));
    float*          bias3  = (float*)(ws + alloc(3 * DHID * 4));
    unsigned short* h1b    = (unsigned short*)(ws + alloc((size_t)NTOK * DHID * 2));
    float*          h2f    = (float*)(ws + alloc((size_t)NTOK * DHID * 4));
    unsigned short* hb     = (unsigned short*)(ws + alloc((size_t)NTOK * DHID * 2));
    float*          qkvf   = (float*)(ws + alloc((size_t)NTOK * 3 * DHID * 4));
    unsigned short* qbuf   = (unsigned short*)(ws + alloc((size_t)NHEAD * NTOK * DHEAD * 2));
    unsigned short* kbuf   = (unsigned short*)(ws + alloc((size_t)NHEAD * NTOK * DHEAD * 2));
    unsigned short* vbuf   = (unsigned short*)(ws + alloc((size_t)NHEAD * NTOK * DHEAD * 2));
    unsigned short* obuf   = (unsigned short*)(ws + alloc((size_t)NTOK * DHID * 2));

    auto cvt = [&](const float* src, unsigned short* dst, int n) {
        f2bf_kernel<<<(n / 4 + 255) / 256, 256, 0, stream>>>(src, dst, n);
    };
    cvt(x, xb, NTOK * CIMG);
    cvt(w1, w1b, CIMG * DHID); // only the image slice (first 3072 rows)
    cvt(w2, w2b, DHID * DHID);
    cvt(wqkv, wqkvb, DHID * 3 * DHID);
    cvt(wproj, wprojb, DHID * CIMG);

    bias_prep_kernel<<<3, 256, 0, stream>>>(cond, w1, b1, bias3);

    const int MT128 = (NTOK + 127) / 128; // 21
    const int MT64  = (NTOK + 63) / 64;   // 42
    // GEMM1: h1 = gelu(x @ w1[:3072] + bias3[t]) -> bf16
    gemm_bf16_kernel<0, 0><<<dim3(DHID / 128, MT128), 256, 0, stream>>>(
        xb, w1b, nullptr, bias3, h1b, NTOK, DHID, CIMG);
    // GEMM2: h2 = h1 @ w2 + b2 -> f32
    gemm_bf16_kernel<1, 0><<<dim3(DHID / 128, MT128), 256, 0, stream>>>(
        h1b, w2b, b2, nullptr, h2f, NTOK, DHID, DHID);
    // LayerNorm -> bf16
    ln_kernel<<<NTOK, 256, 0, stream>>>(h2f, ln_g, ln_b, hb);
    // QKV GEMM -> f32
    gemm_bf16_kernel<1, 0><<<dim3(3 * DHID / 128, MT128), 256, 0, stream>>>(
        hb, wqkvb, bqkv, nullptr, qkvf, NTOK, 3 * DHID, DHID);
    // split + RMS-norm q/k -> head-major bf16
    qkv_post_kernel<<<dim3(NHEAD, NTOK), 64, 0, stream>>>(qkvf, qn_w, kn_w, qbuf, kbuf, vbuf);
    // flash attention
    flash_attn_kernel<<<dim3(MT64, NHEAD), 128, 0, stream>>>(qbuf, kbuf, vbuf, obuf);
    // projection + residual, output row m reads token n=(m%880)*3 + m/880
    gemm_bf16_kernel<2, 1><<<dim3(CIMG / 128, MT128), 256, 0, stream>>>(
        obuf, wprojb, bproj, x, out, NTOK, CIMG, DHID);
    (void)in_sizes; (void)n_in; (void)out_size; (void)ws_size;
}
